// Qwen3Attention_1967095021986
// MI455X (gfx1250) — matrix operand order repaired and checked by reading, not compiled, not run on hardware
//
#include <hip/hip_runtime.h>

typedef _Float16 hv8  __attribute__((ext_vector_type(8)));
typedef _Float16 hv16 __attribute__((ext_vector_type(16)));
typedef float    fv8  __attribute__((ext_vector_type(8)));
typedef unsigned int u32x4 __attribute__((ext_vector_type(4)));
typedef int      i32x4 __attribute__((ext_vector_type(4)));
typedef int      i32x8 __attribute__((ext_vector_type(8)));

#define BB    2
#define SS    1024
#define HIDD  4096
#define NHQ   32
#define NKVH  8
#define HDD   128
#define GSZ   128
#define MTOT  (BB*SS)       // 2048
#define REPF  (NHQ/NKVH)    // 4

#if defined(__has_builtin)
#if __has_builtin(__builtin_amdgcn_tensor_load_to_lds) && __has_builtin(__builtin_amdgcn_s_wait_tensorcnt)
#define HAVE_TDM 1
#endif
#if __has_builtin(__builtin_amdgcn_global_load_async_to_lds_b128)
#define HAVE_ASYNC 1
#endif
#if __has_builtin(__builtin_amdgcn_s_wait_asynccnt)
#define HAVE_ASYNC_WAIT 1
#endif
#endif

// ---------------------------------------------------------------------------
// TDM 2D tile load: global (row-major, 2-byte elems, row stride = stride_d0
// elements) -> LDS contiguous [tile_d1][tile_d0]. D# per ISA 8.3/8.4.
// This toolchain exposes the 6-arg builtin (g0, g1, g2, g3, g4, cpol).
// ---------------------------------------------------------------------------
#if defined(HAVE_TDM)
__device__ __forceinline__ void tdm_load_2d(unsigned lds_byte_addr, const void* gaddr,
                                            unsigned tile_d0, unsigned tile_d1,
                                            unsigned tensor_d0, unsigned tensor_d1,
                                            unsigned stride_d0) {
  unsigned long long ga = (unsigned long long)(uintptr_t)gaddr;
  u32x4 g0;
  g0[0] = 1u;                                          // count=1, user descriptor
  g0[1] = lds_byte_addr;                               // lds_addr
  g0[2] = (unsigned)(ga & 0xFFFFFFFFu);                // global_addr[31:0]
  g0[3] = (unsigned)((ga >> 32) & 0x01FFFFFFu) | 0x80000000u; // ga[56:32] | type=2
  i32x8 g1;
  g1[0] = (int)(1u << 16);                             // data_size=1 (2B), wg_mask=0
  g1[1] = (int)((tensor_d0 & 0xFFFFu) << 16);          // tensor_dim0[15:0]
  g1[2] = (int)((tensor_d0 >> 16) | ((tensor_d1 & 0xFFFFu) << 16)); // td0 hi | td1 lo
  g1[3] = (int)((tensor_d1 >> 16) | (tile_d0 << 16));  // td1 hi | tile_dim0
  g1[4] = (int)(tile_d1 & 0xFFFFu);                    // tile_dim1, tile_dim2=0
  g1[5] = (int)stride_d0;                              // tensor_dim0_stride[31:0]
  g1[6] = 0;                                           // stride hi | dim1_stride lo
  g1[7] = 0;                                           // dim1_stride hi
  i32x4 gz4 = {0, 0, 0, 0};
  i32x8 gz8 = {0, 0, 0, 0, 0, 0, 0, 0};
  __builtin_amdgcn_tensor_load_to_lds(g0, g1, gz4, gz4, gz8, 0);
}
#endif

// ---------------------------------------------------------------------------
// 1) GPTQ int4 dequant -> f16, stored transposed Wt[N][K]
// ---------------------------------------------------------------------------
__global__ __launch_bounds__(256) void dequant_kernel(
    const int* __restrict__ qw, const int* __restrict__ zeros,
    const float* __restrict__ scales, _Float16* __restrict__ wt,
    int in_f, int out_f) {
  int idx = blockIdx.x * blockDim.x + threadIdx.x;
  int in8 = in_f >> 3;
  if (idx >= in8 * out_f) return;
  int c  = idx % out_f;
  int r8 = idx / out_f;
  int g  = (r8 * 8) / GSZ;
  float z  = (float)zeros[(size_t)g * out_f + c];
  float sc = scales[(size_t)g * out_f + c];
  unsigned int packed = (unsigned int)qw[idx];
  hv8 out;
#pragma unroll
  for (int j = 0; j < 8; ++j) {
    float q = (float)((packed >> (4 * j)) & 0xF);
    out[j] = (_Float16)((q - z) * sc);
  }
  *(hv8*)(wt + (size_t)c * in_f + r8 * 8) = out;
}

// ---------------------------------------------------------------------------
// 2) f32 -> f16 conversion (activations)
// ---------------------------------------------------------------------------
__global__ __launch_bounds__(256) void cvt_f16_kernel(
    const float* __restrict__ x, _Float16* __restrict__ y, int n) {
  int i = blockIdx.x * blockDim.x + threadIdx.x;
  if (i < n) y[i] = (_Float16)x[i];
}

// ---------------------------------------------------------------------------
// 3) WMMA GEMM: C[M][N] = A[M][K] x Bt[N][K]^T, f16 in / f32 out.
// 256 threads, block tile 128x128, BK=32, double-buffered LDS fed by TDM
// (fallback: async-load-to-LDS, then plain). Each wave: 4x2 WMMA tiles.
// ---------------------------------------------------------------------------
__global__ __launch_bounds__(256) void gemm_wmma_kernel(
    const _Float16* __restrict__ A, const _Float16* __restrict__ Bt,
    float* __restrict__ C, int M, int N, int K) {
  __shared__ _Float16 As[2][128 * 32];
  __shared__ _Float16 Bs[2][128 * 32];

  const int tid    = threadIdx.x;
  const int lane   = tid & 31;
  const int wave   = tid >> 5;
  const int lane16 = lane & 15;
  const int hsel   = lane >> 4;
  const int wm     = (wave >> 2) * 64;   // 0 / 64
  const int wn     = (wave & 3) * 32;    // 0 / 32 / 64 / 96
  const int m_base = blockIdx.y * 128;
  const int n_base = blockIdx.x * 128;

  fv8 acc[4][2];
#pragma unroll
  for (int mt = 0; mt < 4; ++mt)
#pragma unroll
    for (int nt = 0; nt < 2; ++nt) acc[mt][nt] = fv8{};

  // --- tile staging -------------------------------------------------------
  auto stage = [&](int buf, int k0) {
#if defined(HAVE_TDM)
    if (tid < 32) {
      tdm_load_2d((unsigned)(uintptr_t)&As[buf][0], A + (size_t)m_base * K + k0,
                  32u, 128u, (unsigned)K, (unsigned)M, (unsigned)K);
      tdm_load_2d((unsigned)(uintptr_t)&Bs[buf][0], Bt + (size_t)n_base * K + k0,
                  32u, 128u, (unsigned)K, (unsigned)N, (unsigned)K);
    }
#else
#pragma unroll
    for (int i = 0; i < 2; ++i) {
      int e   = (tid + 256 * i) * 8;          // element offset in 128x32 tile
      int row = e >> 5, col = e & 31;
      const _Float16* ga = A  + (size_t)(m_base + row) * K + k0 + col;
      const _Float16* gb = Bt + (size_t)(n_base + row) * K + k0 + col;
#if defined(HAVE_ASYNC)
      __builtin_amdgcn_global_load_async_to_lds_b128(
          (__attribute__((address_space(1))) const void*)(uintptr_t)ga,
          (__attribute__((address_space(3))) void*)(uintptr_t)(unsigned)(uintptr_t)&As[buf][e], 0, 0);
      __builtin_amdgcn_global_load_async_to_lds_b128(
          (__attribute__((address_space(1))) const void*)(uintptr_t)gb,
          (__attribute__((address_space(3))) void*)(uintptr_t)(unsigned)(uintptr_t)&Bs[buf][e], 0, 0);
#else
      *(hv8*)&As[buf][e] = *(const hv8*)ga;
      *(hv8*)&Bs[buf][e] = *(const hv8*)gb;
#endif
    }
#endif
  };

  stage(0, 0);
  const int nk = K >> 5;
  for (int ki = 0; ki < nk; ++ki) {
    const int  buf      = ki & 1;
    const bool has_next = (ki + 1) < nk;
    if (has_next) stage(buf ^ 1, (ki + 1) << 5);

    // wait for the current tile's DMA (the newest ops belong to the next tile)
#if defined(HAVE_TDM)
    if (tid < 32) {
      if (has_next) __builtin_amdgcn_s_wait_tensorcnt(2);
      else          __builtin_amdgcn_s_wait_tensorcnt(0);
    }
#elif defined(HAVE_ASYNC) && defined(HAVE_ASYNC_WAIT)
    if (has_next) __builtin_amdgcn_s_wait_asynccnt(4);
    else          __builtin_amdgcn_s_wait_asynccnt(0);
#endif
    __syncthreads();

    // B fragments (lane = col n, K = 8*hsel + j and 16 + 8*hsel + j)
    hv16 bfrag[2];
#pragma unroll
    for (int nt = 0; nt < 2; ++nt) {
      const _Float16* br = &Bs[buf][(wn + nt * 16 + lane16) * 32 + 8 * hsel];
      hv8 lo = *(const hv8*)br;
      hv8 hi = *(const hv8*)(br + 16);
#pragma unroll
      for (int j = 0; j < 8; ++j) { bfrag[nt][j] = lo[j]; bfrag[nt][j + 8] = hi[j]; }
    }
#pragma unroll
    for (int mt = 0; mt < 4; ++mt) {
      const _Float16* ar = &As[buf][(wm + mt * 16 + lane16) * 32];
      hv8 lo = *(const hv8*)(ar + 8 * hsel);
      hv8 hi = *(const hv8*)(ar + 16 + 8 * hsel);
      hv16 af;
#pragma unroll
      for (int j = 0; j < 8; ++j) { af[j] = lo[j]; af[j + 8] = hi[j]; }
#pragma unroll
      for (int nt = 0; nt < 2; ++nt)
        acc[mt][nt] = __builtin_amdgcn_wmma_f32_16x16x32_f16(
            false, af, false, bfrag[nt], (short)0, acc[mt][nt], false, false);
    }
    __syncthreads();
  }

  // epilogue: C/D layout -> row = base + r + 8*hsel, col = base + lane16
#pragma unroll
  for (int mt = 0; mt < 4; ++mt)
#pragma unroll
    for (int nt = 0; nt < 2; ++nt)
#pragma unroll
      for (int r = 0; r < 8; ++r)
        C[(size_t)(m_base + wm + mt * 16 + r + 8 * hsel) * N +
          n_base + wn + nt * 16 + lane16] = acc[mt][nt][r];
}

// ---------------------------------------------------------------------------
// 4) Fused per-head RMSNorm + RoPE + relayout, one wave per (token, head)
// ---------------------------------------------------------------------------
__global__ __launch_bounds__(32) void rms_rope_kernel(
    const float* __restrict__ Cq, const float* __restrict__ Ck,
    const float* __restrict__ Cv, const float* __restrict__ cosb,
    const float* __restrict__ sinb, const float* __restrict__ qnw,
    const float* __restrict__ knw, _Float16* __restrict__ Qh,
    _Float16* __restrict__ Kh, _Float16* __restrict__ Vt) {
  int m    = blockIdx.x;
  int slot = blockIdx.y;
  int lane = threadIdx.x;
  int b = m >> 10, s = m & (SS - 1);

  const float* src;
  const float* nw = nullptr;
  bool is_v = false;
  int h;
  size_t dst_base = 0;
  if (slot < NHQ) {
    h = slot;
    src = Cq + (size_t)m * NHQ * HDD + (size_t)h * HDD;
    dst_base = (((size_t)(b * NHQ + h)) * SS + s) * HDD;
    nw = qnw;
  } else if (slot < NHQ + NKVH) {
    h = slot - NHQ;
    src = Ck + (size_t)m * NKVH * HDD + (size_t)h * HDD;
    dst_base = (((size_t)(b * NKVH + h)) * SS + s) * HDD;
    nw = knw;
  } else {
    is_v = true;
    h = slot - NHQ - NKVH;
    src = Cv + (size_t)m * NKVH * HDD + (size_t)h * HDD;
  }

  float4 v4 = *(const float4*)(src + lane * 4);
  float x[4] = {v4.x, v4.y, v4.z, v4.w};

  if (!is_v) {
    float ss2 = x[0]*x[0] + x[1]*x[1] + x[2]*x[2] + x[3]*x[3];
#pragma unroll
    for (int off = 16; off > 0; off >>= 1) ss2 += __shfl_xor(ss2, off, 32);
    float inv = rsqrtf(ss2 / (float)HDD + 1e-6f);
#pragma unroll
    for (int j = 0; j < 4; ++j) x[j] = x[j] * inv * nw[lane * 4 + j];
    float p[4];
#pragma unroll
    for (int j = 0; j < 4; ++j) p[j] = __shfl_xor(x[j], 16, 32);
    const float* cb = cosb + ((size_t)b * SS + s) * HDD + lane * 4;
    const float* sb = sinb + ((size_t)b * SS + s) * HDD + lane * 4;
#pragma unroll
    for (int j = 0; j < 4; ++j) {
      float rot = (lane < 16) ? -p[j] : p[j];
      x[j] = x[j] * cb[j] + rot * sb[j];
    }
    _Float16* dst = (slot < NHQ ? Qh : Kh) + dst_base + lane * 4;
#pragma unroll
    for (int j = 0; j < 4; ++j) dst[j] = (_Float16)x[j];
  } else {
#pragma unroll
    for (int j = 0; j < 4; ++j) {
      int d = lane * 4 + j;
      Vt[(((size_t)(b * NKVH + h)) * HDD + d) * SS + s] = (_Float16)x[j];
    }
  }
}

// ---------------------------------------------------------------------------
// 5) Flash-style causal attention, one wave per (b, h, 16-row q tile).
// ---------------------------------------------------------------------------
__global__ __launch_bounds__(32) void flash_attn_kernel(
    const _Float16* __restrict__ Qh,   // [B][NH][S][HD]
    const _Float16* __restrict__ Kh,   // [B][NKV][S][HD]
    const _Float16* __restrict__ Vt,   // [B][NKV][HD][S]
    _Float16* __restrict__ Oh) {       // [B][S][NH*HD]
  __shared__ float    s_sc[16 * 32];
  __shared__ _Float16 s_p[16 * 32];
  __shared__ float    s_m[16], s_l[16], s_al[16];

  const int lane   = threadIdx.x;
  const int lane16 = lane & 15;
  const int hsel   = lane >> 4;
  const int q0 = blockIdx.x * 16;
  const int h  = blockIdx.y;
  const int b  = blockIdx.z;
  const int hk = h / REPF;
  const float scale = 0.08838834764831845f;   // 1/sqrt(128)

  const _Float16* qrow  = Qh + (((size_t)(b * NHQ + h)) * SS + q0 + lane16) * HDD;
  const _Float16* kbase = Kh + ((size_t)(b * NKVH + hk)) * SS * HDD;
  const _Float16* vbase = Vt + ((size_t)(b * NKVH + hk)) * HDD * SS;

  hv16 qf[4];
#pragma unroll
  for (int c = 0; c < 4; ++c) {
    hv8 lo = *(const hv8*)(qrow + 32 * c + 8 * hsel);
    hv8 hi = *(const hv8*)(qrow + 32 * c + 16 + 8 * hsel);
#pragma unroll
    for (int j = 0; j < 8; ++j) { qf[c][j] = lo[j]; qf[c][j + 8] = hi[j]; }
  }

  fv8 o[8];
#pragma unroll
  for (int t = 0; t < 8; ++t) o[t] = fv8{};

  if (lane < 16) { s_m[lane] = -3.0e38f; s_l[lane] = 0.0f; }
  __syncthreads();

  const int kend = q0 + 16;
  for (int kb = 0; kb < kend; kb += 32) {
#pragma unroll
    for (int jt = 0; jt < 2; ++jt) {
      fv8 sacc = {};
      const _Float16* krow = kbase + (size_t)(kb + 16 * jt + lane16) * HDD;
#pragma unroll
      for (int c = 0; c < 4; ++c) {
        hv8 lo = *(const hv8*)(krow + 32 * c + 8 * hsel);
        hv8 hi = *(const hv8*)(krow + 32 * c + 8 * hsel + 16);
        hv16 bf;
#pragma unroll
        for (int j = 0; j < 8; ++j) { bf[j] = lo[j]; bf[j + 8] = hi[j]; }
        sacc = __builtin_amdgcn_wmma_f32_16x16x32_f16(false, qf[c], false, bf,
                                                      (short)0, sacc, false, false);
      }
#pragma unroll
      for (int r = 0; r < 8; ++r) {
        int row = r + 8 * hsel;
        int key = kb + 16 * jt + lane16;
        float sv = sacc[r] * scale;
        if (key > q0 + row) sv = -3.0e38f;
        s_sc[row * 32 + 16 * jt + lane16] = sv;
      }
    }
    __syncthreads();

    if (lane < 16) {
      int row = lane;
      float mold = s_m[row];
      float mnew = mold;
#pragma unroll
      for (int j = 0; j < 32; ++j) mnew = fmaxf(mnew, s_sc[row * 32 + j]);
      float alpha = __expf(mold - mnew);
      float lsum = 0.0f;
#pragma unroll
      for (int j = 0; j < 32; ++j) {
        float pexp = __expf(s_sc[row * 32 + j] - mnew);
        s_p[row * 32 + j] = (_Float16)pexp;
        lsum += pexp;
      }
      s_m[row]  = mnew;
      s_l[row]  = s_l[row] * alpha + lsum;
      s_al[row] = alpha;
    }
    __syncthreads();

    hv16 pf;
#pragma unroll
    for (int j = 0; j < 16; ++j) {
      int kk = ((j < 8) ? 0 : 16) + (j & 7) + 8 * hsel;
      pf[j] = s_p[lane16 * 32 + kk];
    }
#pragma unroll
    for (int t = 0; t < 8; ++t) {
#pragma unroll
      for (int r = 0; r < 8; ++r) o[t][r] *= s_al[r + 8 * hsel];
      const _Float16* vrow = vbase + (size_t)(16 * t + lane16) * SS + kb + 8 * hsel;
      hv8 lo = *(const hv8*)(vrow);
      hv8 hi = *(const hv8*)(vrow + 16);
      hv16 vf;
#pragma unroll
      for (int j = 0; j < 8; ++j) { vf[j] = lo[j]; vf[j + 8] = hi[j]; }
      o[t] = __builtin_amdgcn_wmma_f32_16x16x32_f16(false, pf, false, vf,
                                                    (short)0, o[t], false, false);
    }
    __syncthreads();
  }

#pragma unroll
  for (int r = 0; r < 8; ++r) {
    int row = r + 8 * hsel;
    float linv = 1.0f / s_l[row];
#pragma unroll
    for (int t = 0; t < 8; ++t) {
      size_t oi = ((size_t)b * SS + q0 + row) * HIDD + (size_t)h * HDD + 16 * t + lane16;
      Oh[oi] = (_Float16)(o[t][r] * linv);
    }
  }
}

// ---------------------------------------------------------------------------
// Host-side orchestration
// ---------------------------------------------------------------------------
extern "C" void kernel_launch(void* const* d_in, const int* in_sizes, int n_in,
                              void* d_out, int out_size, void* d_ws, size_t ws_size,
                              hipStream_t stream) {
  const float* hidden = (const float*)d_in[0];
  const float* cosb   = (const float*)d_in[1];
  const float* sinb   = (const float*)d_in[2];
  const int*   qw_q   = (const int*)d_in[3];
  const int*   zr_q   = (const int*)d_in[4];
  const float* sc_q   = (const float*)d_in[5];
  const int*   qw_k   = (const int*)d_in[6];
  const int*   zr_k   = (const int*)d_in[7];
  const float* sc_k   = (const float*)d_in[8];
  const int*   qw_v   = (const int*)d_in[9];
  const int*   zr_v   = (const int*)d_in[10];
  const float* sc_v   = (const float*)d_in[11];
  const int*   qw_o   = (const int*)d_in[12];
  const int*   zr_o   = (const int*)d_in[13];
  const float* sc_o   = (const float*)d_in[14];
  const float* qnw    = (const float*)d_in[15];
  const float* knw    = (const float*)d_in[16];

  char* ws = (char*)d_ws;
  auto carve = [&](size_t bytes) -> char* {
    char* p = ws;
    ws += (bytes + 255) & ~(size_t)255;
    return p;
  };

  _Float16* Xh  = (_Float16*)carve((size_t)MTOT * HIDD * 2);
  _Float16* WtQ = (_Float16*)carve((size_t)HIDD * HIDD * 2);
  _Float16* WtK = (_Float16*)carve((size_t)(NKVH*HDD) * HIDD * 2);
  _Float16* WtV = (_Float16*)carve((size_t)(NKVH*HDD) * HIDD * 2);
  _Float16* WtO = (_Float16*)carve((size_t)HIDD * HIDD * 2);
  float*    Cq  = (float*)carve((size_t)MTOT * NHQ  * HDD * 4);
  float*    Ck  = (float*)carve((size_t)MTOT * NKVH * HDD * 4);
  float*    Cv  = (float*)carve((size_t)MTOT * NKVH * HDD * 4);
  _Float16* Qh  = (_Float16*)carve((size_t)BB * NHQ  * SS * HDD * 2);
  _Float16* Kh  = (_Float16*)carve((size_t)BB * NKVH * SS * HDD * 2);
  _Float16* Vt  = (_Float16*)carve((size_t)BB * NKVH * HDD * SS * 2);
  _Float16* Oh  = (_Float16*)carve((size_t)MTOT * HIDD * 2);

  {
    int cntQ = (HIDD / 8) * (NHQ * HDD);
    dequant_kernel<<<(cntQ + 255) / 256, 256, 0, stream>>>(qw_q, zr_q, sc_q, WtQ, HIDD, NHQ * HDD);
    int cntK = (HIDD / 8) * (NKVH * HDD);
    dequant_kernel<<<(cntK + 255) / 256, 256, 0, stream>>>(qw_k, zr_k, sc_k, WtK, HIDD, NKVH * HDD);
    dequant_kernel<<<(cntK + 255) / 256, 256, 0, stream>>>(qw_v, zr_v, sc_v, WtV, HIDD, NKVH * HDD);
    int cntO = ((NHQ * HDD) / 8) * HIDD;
    dequant_kernel<<<(cntO + 255) / 256, 256, 0, stream>>>(qw_o, zr_o, sc_o, WtO, NHQ * HDD, HIDD);
  }

  {
    int n = MTOT * HIDD;
    cvt_f16_kernel<<<(n + 255) / 256, 256, 0, stream>>>(hidden, Xh, n);
  }

  gemm_wmma_kernel<<<dim3((NHQ * HDD) / 128, MTOT / 128), 256, 0, stream>>>(Xh, WtQ, Cq, MTOT, NHQ * HDD, HIDD);
  gemm_wmma_kernel<<<dim3((NKVH * HDD) / 128, MTOT / 128), 256, 0, stream>>>(Xh, WtK, Ck, MTOT, NKVH * HDD, HIDD);
  gemm_wmma_kernel<<<dim3((NKVH * HDD) / 128, MTOT / 128), 256, 0, stream>>>(Xh, WtV, Cv, MTOT, NKVH * HDD, HIDD);

  rms_rope_kernel<<<dim3(MTOT, NHQ + 2 * NKVH), 32, 0, stream>>>(
      Cq, Ck, Cv, cosb, sinb, qnw, knw, Qh, Kh, Vt);

  flash_attn_kernel<<<dim3(SS / 16, NHQ, BB), 32, 0, stream>>>(Qh, Kh, Vt, Oh);

  gemm_wmma_kernel<<<dim3(HIDD / 128, MTOT / 128), 256, 0, stream>>>(
      Oh, WtO, (float*)d_out, MTOT, HIDD, HIDD);

  (void)in_sizes; (void)n_in; (void)out_size; (void)ws_size;
}
